// FP8DeQuantLinear_27427661152808
// MI455X (gfx1250) — compile-verified, matrix operand order checked by reading, not run on hardware
//
#include <hip/hip_runtime.h>
#include <hip/hip_bf16.h>

// ---------------------------------------------------------------------------
// FP8-dequant linear:  out[b,s,o] = sum_k x[b,s,k] * (W[o,k]*scale[o]) + bias[o]
// M = B*S = 4096, K = IN = 4096, N = OUT = 11008  (exactly tiled by 128/128/32)
//
// bf16 WMMA (v_wmma_f32_16x16x32_bf16) with fp32 accumulation.
// - dequant + fp32->bf16 (v_cvt_pk_bf16_f32) fused into global->LDS staging
// - double-buffered LDS: 1 barrier per K-step, global loads overlap WMMA
// - non-temporal out stores keep the 180MB weight resident in the 192MB L2
// ---------------------------------------------------------------------------

typedef __attribute__((ext_vector_type(16))) __bf16        v16bf;
typedef __attribute__((ext_vector_type(2)))  __bf16        v2bf;
typedef __attribute__((ext_vector_type(8)))  float         v8f;
typedef __attribute__((ext_vector_type(4)))  unsigned int  u32x4;

#define TILE_M 128
#define TILE_N 128
#define TILE_K 32
#define LDS_STRIDE 40   // bf16 elements per row: 80B pitch, 16B-aligned chunks

__device__ __forceinline__ unsigned int pack2_bf16(float lo, float hi) {
#if __has_builtin(__builtin_amdgcn_cvt_pk_bf16_f32)
    v2bf p = __builtin_amdgcn_cvt_pk_bf16_f32(lo, hi);   // 1 VALU op / 2 elems
    return __builtin_bit_cast(unsigned int, p);
#else
    v2bf p;
    p[0] = (__bf16)lo;   // backend picks hw cvt if present
    p[1] = (__bf16)hi;
    return __builtin_bit_cast(unsigned int, p);
#endif
}

union FragBF16 {
    u32x4 u[2];   // two 16-byte LDS chunks
    v16bf v;      // WMMA operand view
};

__global__ __launch_bounds__(256)
void fp8_dequant_linear_wmma(const float* __restrict__ x,
                             const float* __restrict__ w,
                             const float* __restrict__ wscale,
                             const float* __restrict__ bias,
                             float* __restrict__ out,
                             int M, int N, int K) {
    // double-buffered tiles: 2 * (10240 + 10240) B = 40 KB of the 320 KB WGP LDS
    __shared__ unsigned short lds_a[2][TILE_M * LDS_STRIDE];
    __shared__ unsigned short lds_b[2][TILE_N * LDS_STRIDE];

    const int tid  = threadIdx.x;
    const int lane = tid & 31;
    const int wave = tid >> 5;
    const int wm   = wave & 1;    // 2 wave rows -> 64 M-rows each
    const int wn   = wave >> 1;   // 4 wave cols -> 32 N-cols each

    const int m0 = blockIdx.y * TILE_M;
    const int n0 = blockIdx.x * TILE_N;

    // Global->LDS staging ownership: thread -> (row 0..127, 16-wide K segment)
    const int srow = tid >> 1;
    const int sseg = tid & 1;

    const float* xrow = x + (size_t)(m0 + srow) * K + sseg * 16;
    const float* wrow = w + (size_t)(n0 + srow) * K + sseg * 16;
    const float  scal = wscale[n0 + srow];

    float ra[16], rb[16];

    auto stage_global = [&](int k0) {
        const float4* pa = reinterpret_cast<const float4*>(xrow + k0);
        const float4* pb = reinterpret_cast<const float4*>(wrow + k0);
#pragma unroll
        for (int j = 0; j < 4; ++j) {
            float4 a4 = pa[j];
            float4 b4 = pb[j];
            ra[4*j+0] = a4.x; ra[4*j+1] = a4.y; ra[4*j+2] = a4.z; ra[4*j+3] = a4.w;
            rb[4*j+0] = b4.x; rb[4*j+1] = b4.y; rb[4*j+2] = b4.z; rb[4*j+3] = b4.w;
        }
    };

    auto commit_lds = [&](int buf) {
        unsigned int wa[8], wb[8];
#pragma unroll
        for (int i = 0; i < 8; ++i) {
            wa[i] = pack2_bf16(ra[2*i], ra[2*i+1]);
            wb[i] = pack2_bf16(rb[2*i] * scal, rb[2*i+1] * scal);  // fused dequant
        }
        u32x4* da = reinterpret_cast<u32x4*>(&lds_a[buf][srow * LDS_STRIDE + sseg * 16]);
        u32x4* db = reinterpret_cast<u32x4*>(&lds_b[buf][srow * LDS_STRIDE + sseg * 16]);
        const u32x4* sa = reinterpret_cast<const u32x4*>(wa);
        const u32x4* sb = reinterpret_cast<const u32x4*>(wb);
        da[0] = sa[0]; da[1] = sa[1];
        db[0] = sb[0]; db[1] = sb[1];
    };

    // accumulators: 4 (M) x 2 (N) tiles of 16x16 f32 -> 8 VGPRs each (64 total)
    v8f zero;
#pragma unroll
    for (int e = 0; e < 8; ++e) zero[e] = 0.0f;
    v8f acc[4][2];
#pragma unroll
    for (int i = 0; i < 4; ++i)
#pragma unroll
        for (int j = 0; j < 2; ++j) acc[i][j] = zero;

    const int lm = lane & 15;   // row/col within 16x16 tile
    const int kh = lane >> 4;   // K-half selector per ISA 16-bit A/B layout

    // Prologue: fill buffer 0
    stage_global(0);
    commit_lds(0);
    __syncthreads();

    int cur = 0;
    for (int k0 = 0; k0 < K; k0 += TILE_K) {
        const bool more = (k0 + TILE_K) < K;
        if (more) stage_global(k0 + TILE_K);   // VMEM issues under WMMA below

        FragBF16 afrag[4], bfrag[2];
#pragma unroll
        for (int mt = 0; mt < 4; ++mt) {
            const unsigned short* p = &lds_a[cur][(wm * 64 + mt * 16 + lm) * LDS_STRIDE];
            // lanes 0-15: K 0-7 / 16-23 ; lanes 16-31: K 8-15 / 24-31
            afrag[mt].u[0] = *reinterpret_cast<const u32x4*>(p + kh * 8);
            afrag[mt].u[1] = *reinterpret_cast<const u32x4*>(p + 16 + kh * 8);
        }
#pragma unroll
        for (int nt = 0; nt < 2; ++nt) {
            const unsigned short* p = &lds_b[cur][(wn * 32 + nt * 16 + lm) * LDS_STRIDE];
            bfrag[nt].u[0] = *reinterpret_cast<const u32x4*>(p + kh * 8);
            bfrag[nt].u[1] = *reinterpret_cast<const u32x4*>(p + 16 + kh * 8);
        }

#pragma unroll
        for (int mt = 0; mt < 4; ++mt)
#pragma unroll
            for (int nt = 0; nt < 2; ++nt)
                acc[mt][nt] = __builtin_amdgcn_wmma_f32_16x16x32_bf16(
                    /*neg_a=*/false, afrag[mt].v,
                    /*neg_b=*/false, bfrag[nt].v,
                    /*c_mod=*/(short)0, acc[mt][nt],
                    /*reuse_a=*/false, /*reuse_b=*/false);

        if (more) commit_lds(cur ^ 1);   // write next buffer while cur is read
        __syncthreads();                 // single barrier per K-step
        cur ^= 1;
    }

    // Epilogue: C/D layout -> VGPR r holds M = r + 8*kh, N = lane&15
#pragma unroll
    for (int nt = 0; nt < 2; ++nt) {
        const int n  = n0 + wn * 32 + nt * 16 + lm;
        const float bv = bias[n];
#pragma unroll
        for (int mt = 0; mt < 4; ++mt) {
            const int mbase = m0 + wm * 64 + mt * 16 + kh * 8;
#pragma unroll
            for (int r = 0; r < 8; ++r) {
                // streamed once -> non-temporal, don't evict weight from L2
                __builtin_nontemporal_store(acc[mt][nt][r] + bv,
                                            &out[(size_t)(mbase + r) * N + n]);
            }
        }
    }
}

extern "C" void kernel_launch(void* const* d_in, const int* in_sizes, int n_in,
                              void* d_out, int out_size, void* d_ws, size_t ws_size,
                              hipStream_t stream) {
    const float* x    = (const float*)d_in[0];   // [B,S,IN] fp32
    const float* w    = (const float*)d_in[1];   // [OUT,IN] fp32 (dequant source)
    const float* ws   = (const float*)d_in[2];   // [OUT,1]  fp32 per-channel scale
    const float* bias = (const float*)d_in[3];   // [OUT]    fp32
    float* out = (float*)d_out;                  // [B,S,OUT] fp32

    const int OUT = in_sizes[3];          // 11008
    const int IN  = in_sizes[1] / OUT;    // 4096
    const int M   = in_sizes[0] / IN;     // B*S = 4096

    dim3 grid(OUT / TILE_N, M / TILE_M);  // 86 x 32 (exact tiling for these shapes)
    fp8_dequant_linear_wmma<<<grid, dim3(256), 0, stream>>>(x, w, ws, bias, out,
                                                            M, OUT, IN);
}
